// ResGEM_77867757076596
// MI455X (gfx1250) — compile-verified, matrix operand order checked by reading, not run on hardware
//
#include <hip/hip_runtime.h>
#include <cstdint>
#include <cstddef>

typedef __attribute__((ext_vector_type(16))) _Float16 v16h;
typedef __attribute__((ext_vector_type(8)))  _Float16 v8h;
typedef __attribute__((ext_vector_type(8)))  float    v8f;

#define EPSBN 1e-5f
#define NEGHUGE (-1.0e30f)

static __device__ __forceinline__ float eluf(float v) {
    return v > 0.f ? v : (expf(v) - 1.f);
}

static __device__ __forceinline__ void atomicMaxF(float* p, float v) {
    int* ip = (int*)p;
    int old = __float_as_int(*p);
    while (v > __int_as_float(old)) {
        int assumed = old;
        old = atomicCAS(ip, assumed, __float_as_int(v));
        if (old == assumed) break;
    }
}

// ---------------------------------------------------------------------------
// WMMA GEMM:  C[M,N] = A[M,K] * B[N,K]^T (+ bias[N])
// f32 in global, f16 in LDS, f32 accumulate via v_wmma_f32_16x16x32_f16.
// Block = 256 threads = 8 waves (wave32), block tile 64x64, K-slab 32.
// Wave grid 4x2; each wave owns a 16x32 strip: one A fragment reused
// across two WMMAs (two 16x16 accumulators) per K-slab.
// ---------------------------------------------------------------------------
__global__ __launch_bounds__(256) void gemm_wmma(
    const float* __restrict__ A, int lda,
    const float* __restrict__ B, int ldb,
    const float* __restrict__ bias,
    float* __restrict__ C, int ldc,
    int M, int N, int K)
{
    __shared__ _Float16 As[64 * 40];   // 64 rows x 32 K halves (+8 pad)
    __shared__ _Float16 Bs[64 * 40];   // 64 cols x 32 K halves (+8 pad)

    const int tid  = threadIdx.x;
    const int lane = tid & 31;
    const int wid  = tid >> 5;
    const int wm   = wid & 3;          // 4 waves along M (16 rows each)
    const int wn   = wid >> 2;         // 2 waves along N (32 cols each)
    const int bm   = blockIdx.x * 64;
    const int bn   = blockIdx.y * 64;
    const int mrow = lane & 15;
    const int hi   = lane >> 4;        // K-half / M-half selector

    // staging coordinates (fast path): thread -> (row, 8-col group)
    const int sr  = tid >> 2;          // 0..63
    const int sc  = (tid & 3) * 8;     // 0,8,16,24

    union { v8f v; float f[8]; } acc0, acc1;
    acc0.v = v8f{};
    acc1.v = v8f{};

    const bool alignedA = ((lda & 3) == 0);
    const bool alignedB = ((ldb & 3) == 0);

    for (int k0 = 0; k0 < K; k0 += 32) {
        const bool interior = (bm + 64 <= M) && (bn + 64 <= N) && (k0 + 32 <= K);
        if (interior && alignedA && alignedB) {
            const float* ap = A + (size_t)(bm + sr) * lda + k0 + sc;
            const float* bp = B + (size_t)(bn + sr) * ldb + k0 + sc;
            float4 a0 = *(const float4*)(ap);
            float4 a1 = *(const float4*)(ap + 4);
            float4 b0 = *(const float4*)(bp);
            float4 b1 = *(const float4*)(bp + 4);
            if (k0 + 32 < K) {   // prefetch next K-slab (global_prefetch_b8)
                __builtin_prefetch(ap + 32, 0, 1);
                __builtin_prefetch(bp + 32, 0, 1);
            }
            union { v8h v; _Float16 h[8]; } pa, pb;
            pa.h[0] = (_Float16)a0.x; pa.h[1] = (_Float16)a0.y;
            pa.h[2] = (_Float16)a0.z; pa.h[3] = (_Float16)a0.w;
            pa.h[4] = (_Float16)a1.x; pa.h[5] = (_Float16)a1.y;
            pa.h[6] = (_Float16)a1.z; pa.h[7] = (_Float16)a1.w;
            pb.h[0] = (_Float16)b0.x; pb.h[1] = (_Float16)b0.y;
            pb.h[2] = (_Float16)b0.z; pb.h[3] = (_Float16)b0.w;
            pb.h[4] = (_Float16)b1.x; pb.h[5] = (_Float16)b1.y;
            pb.h[6] = (_Float16)b1.z; pb.h[7] = (_Float16)b1.w;
            *(v8h*)(&As[sr * 40 + sc]) = pa.v;   // 16B aligned
            *(v8h*)(&Bs[sr * 40 + sc]) = pb.v;
        } else {
            #pragma unroll
            for (int u = 0; u < 8; ++u) {
                int e = tid + u * 256;           // 2048 elements
                int r = e >> 5, kk = e & 31;
                int gr = bm + r, gk = k0 + kk;
                float v = (gr < M && gk < K) ? A[(size_t)gr * lda + gk] : 0.f;
                As[r * 40 + kk] = (_Float16)v;
                int gc = bn + r;
                float w = (gc < N && gk < K) ? B[(size_t)gc * ldb + gk] : 0.f;
                Bs[r * 40 + kk] = (_Float16)w;
            }
        }
        __syncthreads();

        union { v16h v; _Float16 h[16]; } afr, bfr0, bfr1;
        #pragma unroll
        for (int vv = 0; vv < 8; ++vv) {
            #pragma unroll
            for (int hh = 0; hh < 2; ++hh) {
                int kk = (vv < 4 ? 2 * vv + hh : 16 + 2 * (vv - 4) + hh) + hi * 8;
                afr.h[2 * vv + hh]  = As[(wm * 16 + mrow) * 40 + kk];
                bfr0.h[2 * vv + hh] = Bs[(wn * 32 + mrow) * 40 + kk];
                bfr1.h[2 * vv + hh] = Bs[(wn * 32 + 16 + mrow) * 40 + kk];
            }
        }
        acc0.v = __builtin_amdgcn_wmma_f32_16x16x32_f16(
            false, afr.v, false, bfr0.v, (short)0, acc0.v, false, false);
        acc1.v = __builtin_amdgcn_wmma_f32_16x16x32_f16(
            false, afr.v, false, bfr1.v, (short)0, acc1.v, false, false);
        __syncthreads();
    }

    const int col0 = bn + wn * 32 + mrow;
    const int col1 = col0 + 16;
    const float bv0 = (bias != nullptr && col0 < N) ? bias[col0] : 0.f;
    const float bv1 = (bias != nullptr && col1 < N) ? bias[col1] : 0.f;
    #pragma unroll
    for (int r = 0; r < 8; ++r) {
        int row = bm + wm * 16 + r + hi * 8;
        if (row < M) {
            if (col0 < N) C[(size_t)row * ldc + col0] = acc0.f[r] + bv0;
            if (col1 < N) C[(size_t)row * ldc + col1] = acc1.f[r] + bv1;
        }
    }
}

// ---------------------------------------------------------------------------
// f32 -> padded f16 matrix copy (zero-pads K to Kpad) for the Gram kernel.
// ---------------------------------------------------------------------------
__global__ void cvt_f16_pad(const float* __restrict__ x, int ld, int C, int N,
                            _Float16* __restrict__ out, int kpad) {
    int t = blockIdx.x * blockDim.x + threadIdx.x;
    if (t >= N * kpad) return;
    int n = t / kpad, c = t % kpad;
    float v = (c < C) ? x[(size_t)n * ld + c] : 0.f;
    out[(size_t)n * kpad + c] = (_Float16)v;
}

// ---------------------------------------------------------------------------
// Gram strip GEMM with ASYNC global->LDS staging (CDNA5 ASYNCcnt path):
//   C[M=gridX*64, N] = Xh[row0+..] * Xh^T, Xh f16 [*, ldh], ldh % 32 == 0.
// All tiles interior by construction -> no guards.  Each of the 256 threads
// issues exactly one GLOBAL_LOAD_ASYNC_TO_LDS_B128 per operand per K-slab
// (16 bytes/lane), then s_wait_asynccnt 0 + barrier before the WMMAs.
// ---------------------------------------------------------------------------
__global__ __launch_bounds__(256) void gram_wmma_async(
    const _Float16* __restrict__ Xh, int ldh, int row0,
    float* __restrict__ C, int N)
{
    __shared__ _Float16 As[64 * 40];
    __shared__ _Float16 Bs[64 * 40];

    const int tid  = threadIdx.x;
    const int lane = tid & 31;
    const int wid  = tid >> 5;
    const int wm   = wid & 3;
    const int wn   = wid >> 2;
    const int bm   = blockIdx.x * 64;
    const int bn   = blockIdx.y * 64;
    const int mrow = lane & 15;
    const int hi   = lane >> 4;

    const int sr  = tid >> 2;          // 0..63
    const int sc  = (tid & 3) * 8;     // 0,8,16,24 halves (16B chunks)

    const unsigned ldsA = (unsigned)(uintptr_t)(&As[sr * 40 + sc]);
    const unsigned ldsB = (unsigned)(uintptr_t)(&Bs[sr * 40 + sc]);

    union { v8f v; float f[8]; } acc0, acc1;
    acc0.v = v8f{};
    acc1.v = v8f{};

    for (int k0 = 0; k0 < ldh; k0 += 32) {
        uint64_t ga = (uint64_t)(const void*)(Xh + (size_t)(row0 + bm + sr) * ldh + k0 + sc);
        uint64_t gb = (uint64_t)(const void*)(Xh + (size_t)(bn + sr) * ldh + k0 + sc);
        asm volatile("global_load_async_to_lds_b128 %0, %1, off"
                     :: "v"(ldsA), "v"(ga) : "memory");
        asm volatile("global_load_async_to_lds_b128 %0, %1, off"
                     :: "v"(ldsB), "v"(gb) : "memory");
        asm volatile("s_wait_asynccnt 0x0" ::: "memory");
        __syncthreads();

        union { v16h v; _Float16 h[16]; } afr, bfr0, bfr1;
        #pragma unroll
        for (int vv = 0; vv < 8; ++vv) {
            #pragma unroll
            for (int hh = 0; hh < 2; ++hh) {
                int kk = (vv < 4 ? 2 * vv + hh : 16 + 2 * (vv - 4) + hh) + hi * 8;
                afr.h[2 * vv + hh]  = As[(wm * 16 + mrow) * 40 + kk];
                bfr0.h[2 * vv + hh] = Bs[(wn * 32 + mrow) * 40 + kk];
                bfr1.h[2 * vv + hh] = Bs[(wn * 32 + 16 + mrow) * 40 + kk];
            }
        }
        acc0.v = __builtin_amdgcn_wmma_f32_16x16x32_f16(
            false, afr.v, false, bfr0.v, (short)0, acc0.v, false, false);
        acc1.v = __builtin_amdgcn_wmma_f32_16x16x32_f16(
            false, afr.v, false, bfr1.v, (short)0, acc1.v, false, false);
        __syncthreads();
    }

    const int col0 = bn + wn * 32 + mrow;
    const int col1 = col0 + 16;
    #pragma unroll
    for (int r = 0; r < 8; ++r) {
        int row = bm + wm * 16 + r + hi * 8;
        C[(size_t)row * N + col0] = acc0.f[r];
        C[(size_t)row * N + col1] = acc1.f[r];
    }
}

// ---------------------------------------------------------------------------
// Utility kernels
// ---------------------------------------------------------------------------
__global__ void fill_f32(float* p, long n, float v) {
    long t = (long)blockIdx.x * blockDim.x + threadIdx.x;
    if (t < n) p[t] = v;
}

__global__ void iota_i32(int* p, int n) {
    int t = blockIdx.x * blockDim.x + threadIdx.x;
    if (t < n) p[t] = t;
}

__global__ void row_sqnorm(const float* __restrict__ x, int ld, int C, int N,
                           float* __restrict__ xx) {
    int i = blockIdx.x * blockDim.x + threadIdx.x;
    if (i >= N) return;
    float s = 0.f;
    for (int c = 0; c < C; ++c) { float v = x[(size_t)i * ld + c]; s += v * v; }
    xx[i] = s;
}

// O1 = c00*Wfirst + c01*Wsecond ; O2 = c10*Wfirst + c11*Wsecond
__global__ void split_combo(const float* __restrict__ W, int o, int ci,
                            float c00, float c01, float c10, float c11,
                            float* __restrict__ O1, float* __restrict__ O2) {
    int t = blockIdx.x * blockDim.x + threadIdx.x;
    if (t >= o * ci) return;
    int r = t / ci, c = t % ci;
    float A = W[(size_t)r * 2 * ci + c];
    float B = W[(size_t)r * 2 * ci + ci + c];
    O1[(size_t)r * ci + c] = c00 * A + c01 * B;
    O2[(size_t)r * ci + c] = c10 * A + c11 * B;
}

__global__ void transpose_mat(const float* __restrict__ in, int rows, int cols,
                              float* __restrict__ out) {
    int t = blockIdx.x * blockDim.x + threadIdx.x;
    if (t >= rows * cols) return;
    int r = t / cols, c = t % cols;
    out[(size_t)c * rows + r] = in[(size_t)r * cols + c];
}

// ---------------------------------------------------------------------------
// kNN: wave-per-row top-k selection over a precomputed Gram strip.
// ---------------------------------------------------------------------------
#define MAXK 32
__global__ __launch_bounds__(32) void knn_select(
    const float* __restrict__ strip, const float* __restrict__ xx,
    int row0, int N, int k, int* __restrict__ idx)
{
    const int r = blockIdx.x;
    const int i = row0 + r;
    const int lane = threadIdx.x;

    float bd[MAXK]; int bi[MAXK];
    for (int t = 0; t < k; ++t) { bd[t] = NEGHUGE; bi[t] = 0; }

    const float xi = xx[i];
    const float* srow = strip + (size_t)r * N;
    for (int j = lane; j < N; j += 32) {
        float d = 2.f * srow[j] - xi - xx[j];
        if (d > bd[k - 1]) {
            int p = k - 1;
            while (p > 0 && bd[p - 1] < d) {
                bd[p] = bd[p - 1]; bi[p] = bi[p - 1]; --p;
            }
            bd[p] = d; bi[p] = j;
        }
    }

    __shared__ float sd[32 * MAXK];
    __shared__ int   si[32 * MAXK];
    __shared__ int   shp[32];
    for (int t = 0; t < k; ++t) {
        sd[lane * MAXK + t] = bd[t];
        si[lane * MAXK + t] = bi[t];
    }
    shp[lane] = 0;
    __syncthreads();

    if (lane == 0) {
        for (int t = 0; t < k; ++t) {
            float best = NEGHUGE; int bl = 0;
            for (int l = 0; l < 32; ++l) {
                int hp = shp[l];
                if (hp < k && sd[l * MAXK + hp] > best) {
                    best = sd[l * MAXK + hp]; bl = l;
                }
            }
            idx[(size_t)i * k + t] = si[bl * MAXK + shp[bl]];
            shp[bl]++;
        }
    }
}

// ---------------------------------------------------------------------------
// FEM stats + fused max/BN/ELU: h[n,kk,o] = Yd[idx[n,kk],o] + Ye[n,o]
// ---------------------------------------------------------------------------
__global__ __launch_bounds__(256) void fem_stats(
    const float* __restrict__ Yd, const float* __restrict__ Ye,
    const int* __restrict__ idx, int N, int k, int outc,
    float* __restrict__ stats)
{
    const int o = blockIdx.x;
    float s = 0.f, ss = 0.f;
    for (int n = threadIdx.x; n < N; n += 256) {
        float ye = Ye[(size_t)n * outc + o];
        const int* ip = idx + (size_t)n * k;
        for (int t = 0; t < k; ++t) {
            float v = Yd[(size_t)ip[t] * outc + o] + ye;
            s += v; ss += v * v;
        }
    }
    __shared__ float rs[256], rq[256];
    rs[threadIdx.x] = s; rq[threadIdx.x] = ss;
    __syncthreads();
    for (int w = 128; w > 0; w >>= 1) {
        if (threadIdx.x < w) {
            rs[threadIdx.x] += rs[threadIdx.x + w];
            rq[threadIdx.x] += rq[threadIdx.x + w];
        }
        __syncthreads();
    }
    if (threadIdx.x == 0) { stats[o] = rs[0]; stats[outc + o] = rq[0]; }
}

__global__ void fem_max_bn(
    const float* __restrict__ Yd, const float* __restrict__ Ye,
    const int* __restrict__ idx, const float* __restrict__ stats,
    int N, int k, int outc, float invcnt, float* __restrict__ out, int ldo)
{
    int t = blockIdx.x * blockDim.x + threadIdx.x;
    if (t >= N * outc) return;
    int n = t / outc, o = t % outc;
    const int* ip = idx + (size_t)n * k;
    float m = NEGHUGE;
    for (int kk = 0; kk < k; ++kk)
        m = fmaxf(m, Yd[(size_t)ip[kk] * outc + o]);
    m += Ye[(size_t)n * outc + o];
    float mean = stats[o] * invcnt;
    float var  = stats[outc + o] * invcnt - mean * mean;
    float v = (m - mean) * rsqrtf(var + EPSBN);
    out[(size_t)n * ldo + o] = eluf(v);
}

// ---------------------------------------------------------------------------
// Generic per-column BN stats (+optional residual) and BN+act apply
// ---------------------------------------------------------------------------
__global__ __launch_bounds__(256) void col_stats(
    const float* __restrict__ x, const float* __restrict__ res,
    int rows, int cols, int ldx, int ldr, float* __restrict__ stats)
{
    const int c = blockIdx.x;
    float s = 0.f, ss = 0.f;
    for (int r = threadIdx.x; r < rows; r += 256) {
        float v = x[(size_t)r * ldx + c];
        if (res) v += res[(size_t)r * ldr + c];
        s += v; ss += v * v;
    }
    __shared__ float rs[256], rq[256];
    rs[threadIdx.x] = s; rq[threadIdx.x] = ss;
    __syncthreads();
    for (int w = 128; w > 0; w >>= 1) {
        if (threadIdx.x < w) {
            rs[threadIdx.x] += rs[threadIdx.x + w];
            rq[threadIdx.x] += rq[threadIdx.x + w];
        }
        __syncthreads();
    }
    if (threadIdx.x == 0) { stats[c] = rs[0]; stats[cols + c] = rq[0]; }
}

__global__ void bn_act(
    const float* __restrict__ x, const float* __restrict__ res,
    const float* __restrict__ stats, int rows, int cols, int ldx, int ldr,
    float invcnt, int elu, float* __restrict__ out, int ldo)
{
    int t = blockIdx.x * blockDim.x + threadIdx.x;
    if (t >= rows * cols) return;
    int r = t / cols, c = t % cols;
    float v = x[(size_t)r * ldx + c];
    if (res) v += res[(size_t)r * ldr + c];
    float mean = stats[c] * invcnt;
    float var  = stats[cols + c] * invcnt - mean * mean;
    v = (v - mean) * rsqrtf(var + EPSBN);
    if (elu) v = eluf(v);
    out[(size_t)r * ldo + c] = v;
}

// ---------------------------------------------------------------------------
// EdgeConv helpers
// ---------------------------------------------------------------------------
__global__ __launch_bounds__(256) void edge_stats(
    const float* __restrict__ P, const float* __restrict__ Q,
    const float* __restrict__ b1, const int* __restrict__ src,
    const int* __restrict__ tgt, int E, int outc, float* __restrict__ stats)
{
    const int o = blockIdx.x;
    const float bb = b1[o];
    float s = 0.f, ss = 0.f;
    for (int e = threadIdx.x; e < E; e += 256) {
        float v = P[(size_t)tgt[e] * outc + o] + Q[(size_t)src[e] * outc + o] + bb;
        s += v; ss += v * v;
    }
    __shared__ float rs[256], rq[256];
    rs[threadIdx.x] = s; rq[threadIdx.x] = ss;
    __syncthreads();
    for (int w = 128; w > 0; w >>= 1) {
        if (threadIdx.x < w) {
            rs[threadIdx.x] += rs[threadIdx.x + w];
            rq[threadIdx.x] += rq[threadIdx.x + w];
        }
        __syncthreads();
    }
    if (threadIdx.x == 0) { stats[o] = rs[0]; stats[outc + o] = rq[0]; }
}

__global__ void edge_h(
    const float* __restrict__ P, const float* __restrict__ Q,
    const float* __restrict__ b1, const int* __restrict__ src,
    const int* __restrict__ tgt, int e0, int ecnt, int outc,
    const float* __restrict__ stats, float invcnt, float* __restrict__ H)
{
    int t = blockIdx.x * blockDim.x + threadIdx.x;
    if (t >= ecnt * outc) return;
    int r = t / outc, o = t % outc;
    int e = e0 + r;
    float v = P[(size_t)tgt[e] * outc + o] + Q[(size_t)src[e] * outc + o] + b1[o];
    float mean = stats[o] * invcnt;
    float var  = stats[outc + o] * invcnt - mean * mean;
    v = (v - mean) * rsqrtf(var + EPSBN);
    H[(size_t)r * outc + o] = eluf(v);
}

__global__ void seg_max(
    const float* __restrict__ H, const int* __restrict__ tgt,
    int e0, int ecnt, int outc, float* __restrict__ out)
{
    int t = blockIdx.x * blockDim.x + threadIdx.x;
    if (t >= ecnt * outc) return;
    int r = t / outc, o = t % outc;
    atomicMaxF(&out[(size_t)tgt[e0 + r] * outc + o], H[(size_t)r * outc + o]);
}

__global__ void fix_neginf(float* p, long n) {
    long t = (long)blockIdx.x * blockDim.x + threadIdx.x;
    if (t < n && p[t] <= -1.0e29f) p[t] = 0.f;
}

// ---------------------------------------------------------------------------
// FeaSt
// ---------------------------------------------------------------------------
__global__ void xu_small(
    const float* __restrict__ x, int ldx, const float* __restrict__ u,
    int N, int Cin, float* __restrict__ U)
{
    int t = blockIdx.x * blockDim.x + threadIdx.x;
    if (t >= N * 8) return;
    int n = t / 8, h = t % 8;
    float s = 0.f;
    for (int c = 0; c < Cin; ++c)
        s += x[(size_t)n * ldx + c] * u[(size_t)c * 8 + h];
    U[(size_t)n * 8 + h] = s;
}

__global__ void feast_edge(
    const float* __restrict__ U, const float* __restrict__ cpar,
    const float* __restrict__ XW, const int* __restrict__ src,
    const int* __restrict__ tgt, int E, float* __restrict__ S,
    float* __restrict__ CNT)
{
    int e = blockIdx.x * blockDim.x + threadIdx.x;
    if (e >= E) return;
    int sj = src[e], ti = tgt[e];
    float q[8];
    float mx = NEGHUGE;
    #pragma unroll
    for (int h = 0; h < 8; ++h) {
        q[h] = U[(size_t)ti * 8 + h] - U[(size_t)sj * 8 + h] + cpar[h];
        mx = fmaxf(mx, q[h]);
    }
    float sum = 0.f;
    #pragma unroll
    for (int h = 0; h < 8; ++h) { q[h] = expf(q[h] - mx); sum += q[h]; }
    float inv = 1.f / sum;
    const float* xw = XW + (size_t)sj * 512;
    for (int c = 0; c < 64; ++c) {
        float m = 0.f;
        #pragma unroll
        for (int h = 0; h < 8; ++h) m += xw[h * 64 + c] * q[h];
        atomicAdd(&S[(size_t)ti * 64 + c], m * inv);
    }
    atomicAdd(&CNT[ti], 1.f);
}

__global__ void feast_finish(
    const float* __restrict__ S, const float* __restrict__ CNT,
    const float* __restrict__ b, int N, float* __restrict__ out)
{
    int t = blockIdx.x * blockDim.x + threadIdx.x;
    if (t >= N * 64) return;
    int n = t / 64, c = t % 64;
    out[(size_t)n * 64 + c] = S[(size_t)n * 64 + c] / fmaxf(CNT[n], 1.f) + b[c];
}

// ---------------------------------------------------------------------------
// Host orchestration
// ---------------------------------------------------------------------------
extern "C" void kernel_launch(void* const* d_in, const int* in_sizes, int n_in,
                              void* d_out, int out_size, void* d_ws, size_t ws_size,
                              hipStream_t stream)
{
    (void)n_in; (void)ws_size; (void)out_size;
    const int N = in_sizes[0] / 3;
    const int E = in_sizes[1] / 2;
    const float* x_in = (const float*)d_in[0];
    const int* src = (const int*)d_in[1];
    const int* tgt = src + E;

    int pi = 2;
    struct FemP { const float *W1, *W2, *W3, *W4, *fcW, *fcb; } fem[3];
    for (int s = 0; s < 3; ++s) {
        fem[s].W1  = (const float*)d_in[pi++];
        fem[s].W2  = (const float*)d_in[pi++];
        fem[s].W3  = (const float*)d_in[pi++];
        fem[s].W4  = (const float*)d_in[pi++];
        fem[s].fcW = (const float*)d_in[pi++];
        fem[s].fcb = (const float*)d_in[pi++];
    }
    struct EcP { const float *W1, *b1, *W2, *b2; } ec[3];
    for (int s = 0; s < 3; ++s) {
        ec[s].W1 = (const float*)d_in[pi++];
        ec[s].b1 = (const float*)d_in[pi++];
        ec[s].W2 = (const float*)d_in[pi++];
        ec[s].b2 = (const float*)d_in[pi++];
    }
    struct RbP { const float *fc1W, *fc1b, *fc2W, *fc2b, *fW, *fu, *fc, *fb; } rb[4];
    for (int s = 0; s < 4; ++s) {
        rb[s].fc1W = (const float*)d_in[pi++];
        rb[s].fc1b = (const float*)d_in[pi++];
        rb[s].fc2W = (const float*)d_in[pi++];
        rb[s].fc2b = (const float*)d_in[pi++];
        rb[s].fW   = (const float*)d_in[pi++];
        rb[s].fu   = (const float*)d_in[pi++];
        rb[s].fc   = (const float*)d_in[pi++];
        rb[s].fb   = (const float*)d_in[pi++];
    }
    const float* fc1W = (const float*)d_in[pi++];
    const float* fc1b = (const float*)d_in[pi++];
    const float* fc2W = (const float*)d_in[pi++];
    const float* fc2b = (const float*)d_in[pi++];

    char* wsp = (char*)d_ws;
    auto alloc = [&](size_t bytes) -> void* {
        void* p = (void*)wsp;
        wsp += (bytes + 255) & ~(size_t)255;
        return p;
    };
    const int CH  = 512;    // kNN Gram strip rows (16 MB strip -> L2 resident)
    const int ECH = 16384;  // edge chunk (16 MB chunk buffers)

    float*    xx    = (float*)alloc((size_t)N * 4);
    float*    strip = (float*)alloc((size_t)CH * N * 4);
    int*      idx   = (int*)  alloc((size_t)N * 32 * 4);
    _Float16* Xh    = (_Float16*)alloc((size_t)N * 128 * 2);
    float*    Hfem  = (float*)alloc((size_t)N * 512 * 4);
    float*    Yd    = (float*)alloc((size_t)N * 256 * 4);
    float*    Ye    = (float*)alloc((size_t)N * 256 * 4);
    float*    stats = (float*)alloc((size_t)2 * 1024 * 4);
    float*    Wt1   = (float*)alloc((size_t)512 * 256 * 4);
    float*    Wt2   = (float*)alloc((size_t)512 * 256 * 4);
    float*    HMAIN = (float*)alloc((size_t)N * 1024 * 4);
    float*    Htmp  = (float*)alloc((size_t)N * 256 * 4);
    float*    P     = (float*)alloc((size_t)N * 256 * 4);
    float*    Q     = (float*)alloc((size_t)N * 256 * 4);
    float*    EO    = (float*)alloc((size_t)N * 256 * 4);
    float*    EA    = (float*)alloc((size_t)N * 256 * 4);
    float*    EB    = (float*)alloc((size_t)N * 256 * 4);
    float*    EH1   = (float*)alloc((size_t)ECH * 256 * 4);
    float*    EH2   = (float*)alloc((size_t)ECH * 256 * 4);
    float*    T1    = (float*)alloc((size_t)N * 64 * 4);
    float*    T1n   = (float*)alloc((size_t)N * 64 * 4);
    float*    XWp   = (float*)alloc((size_t)N * 512 * 4);
    float*    Uf    = (float*)alloc((size_t)N * 8 * 4);
    float*    Sacc  = (float*)alloc((size_t)N * 64 * 4);
    float*    CNT   = (float*)alloc((size_t)N * 4);
    float*    T2    = (float*)alloc((size_t)N * 64 * 4);
    float*    T3    = (float*)alloc((size_t)N * 1024 * 4);
    int*      iota  = (int*)  alloc((size_t)N * 4);

    auto gemm = [&](const float* A, int lda, const float* B, int ldb,
                    const float* bias, float* C, int ldc, int M, int Nn, int K) {
        dim3 g((M + 63) / 64, (Nn + 63) / 64);
        gemm_wmma<<<g, 256, 0, stream>>>(A, lda, B, ldb, bias, C, ldc, M, Nn, K);
    };
    auto grid1 = [](long n) { return (unsigned)((n + 255) / 256); };

    iota_i32<<<grid1(N), 256, 0, stream>>>(iota, N);

    // ================= FEM branches (neighbor sizes 8,16,32) ================
    const int ksz[3] = {8, 16, 32};
    for (int s = 0; s < 3; ++s) {
        const int kk = ksz[s];
        const float* Wl[4] = {fem[s].W1, fem[s].W2, fem[s].W3, fem[s].W4};
        const int cin[4]  = {3, 64, 64, 128};
        const int cout[4] = {64, 64, 128, 256};
        const int coff[4] = {0, 64, 128, 256};

        const float* Xc = x_in; int ldx = 3;
        for (int l = 0; l < 4; ++l) {
            const int Ci = cin[l], Co = cout[l];
            const int Kpad = (Ci + 31) & ~31;
            // kNN on current features: f16 copy + async-staged Gram strips
            row_sqnorm<<<grid1(N), 256, 0, stream>>>(Xc, ldx, Ci, N, xx);
            cvt_f16_pad<<<grid1((long)N * Kpad), 256, 0, stream>>>(
                Xc, ldx, Ci, N, Xh, Kpad);
            for (int r0 = 0; r0 < N; r0 += CH) {
                dim3 gg(CH / 64, N / 64);
                gram_wmma_async<<<gg, 256, 0, stream>>>(Xh, Kpad, r0, strip, N);
                knn_select<<<CH, 32, 0, stream>>>(strip, xx, r0, N, kk, idx);
            }
            // Wd = W[:, :Ci], We = W[:, Ci:] - W[:, :Ci]
            split_combo<<<grid1((long)Co * Ci), 256, 0, stream>>>(
                Wl[l], Co, Ci, 1.f, 0.f, -1.f, 1.f, Wt1, Wt2);
            gemm(Xc, ldx, Wt1, Ci, nullptr, Yd, Co, N, Co, Ci);
            gemm(Xc, ldx, Wt2, Ci, nullptr, Ye, Co, N, Co, Ci);
            fem_stats<<<Co, 256, 0, stream>>>(Yd, Ye, idx, N, kk, Co, stats);
            fem_max_bn<<<grid1((long)N * Co), 256, 0, stream>>>(
                Yd, Ye, idx, stats, N, kk, Co, 1.f / ((float)N * kk),
                Hfem + coff[l], 512);
            Xc = Hfem + coff[l]; ldx = 512;
        }
        gemm(Hfem, 512, fem[s].fcW, 512, fem[s].fcb, Htmp, 256, N, 256, 512);
        col_stats<<<256, 256, 0, stream>>>(Htmp, nullptr, N, 256, 256, 0, stats);
        bn_act<<<grid1((long)N * 256), 256, 0, stream>>>(
            Htmp, nullptr, stats, N, 256, 256, 0, 1.f / N, 1,
            HMAIN + 256 + s * 256, 1024);
    }

    // ================= EdgeConv chain ================
    auto edgeconv = [&](const float* Xin, int ldin, int Ci, int Co,
                        const EcP& pp, float* OUTbuf) {
        split_combo<<<grid1((long)Co * Ci), 256, 0, stream>>>(
            pp.W1, Co, Ci, 1.f, -1.f, 0.f, 1.f, Wt1, Wt2);
        gemm(Xin, ldin, Wt1, Ci, nullptr, P, Co, N, Co, Ci);
        gemm(Xin, ldin, Wt2, Ci, nullptr, Q, Co, N, Co, Ci);
        edge_stats<<<Co, 256, 0, stream>>>(P, Q, pp.b1, src, tgt, E, Co, stats);
        fill_f32<<<grid1((long)N * Co), 256, 0, stream>>>(OUTbuf, (long)N * Co, NEGHUGE);
        for (int e0 = 0; e0 < E; e0 += ECH) {
            int ec_ = (E - e0 < ECH) ? (E - e0) : ECH;
            edge_h<<<grid1((long)ec_ * Co), 256, 0, stream>>>(
                P, Q, pp.b1, src, tgt, e0, ec_, Co, stats, 1.f / E, EH1);
            gemm(EH1, Co, pp.W2, Co, pp.b2, EH2, Co, ec_, Co, Co);
            seg_max<<<grid1((long)ec_ * Co), 256, 0, stream>>>(
                EH2, tgt, e0, ec_, Co, OUTbuf);
        }
        fix_neginf<<<grid1((long)N * Co), 256, 0, stream>>>(OUTbuf, (long)N * Co);
    };

    edgeconv(x_in, 3, 3, 64, ec[0], EO);
    col_stats<<<64, 256, 0, stream>>>(EO, nullptr, N, 64, 64, 0, stats);
    bn_act<<<grid1((long)N * 64), 256, 0, stream>>>(
        EO, nullptr, stats, N, 64, 64, 0, 1.f / N, 1, EA, 64);

    edgeconv(EA, 64, 64, 128, ec[1], EO);
    col_stats<<<128, 256, 0, stream>>>(EO, nullptr, N, 128, 128, 0, stats);
    bn_act<<<grid1((long)N * 128), 256, 0, stream>>>(
        EO, nullptr, stats, N, 128, 128, 0, 1.f / N, 1, EB, 128);

    edgeconv(EB, 128, 128, 256, ec[2], EO);
    col_stats<<<256, 256, 0, stream>>>(EO, nullptr, N, 256, 256, 0, stats);
    bn_act<<<grid1((long)N * 256), 256, 0, stream>>>(
        EO, nullptr, stats, N, 256, 256, 0, 1.f / N, 1, HMAIN, 1024);

    // ================= Residual FeaSt blocks ================
    for (int b = 0; b < 4; ++b) {
        gemm(HMAIN, 1024, rb[b].fc1W, 1024, rb[b].fc1b, T1, 64, N, 64, 1024);
        col_stats<<<64, 256, 0, stream>>>(T1, nullptr, N, 64, 64, 0, stats);
        bn_act<<<grid1((long)N * 64), 256, 0, stream>>>(
            T1, nullptr, stats, N, 64, 64, 0, 1.f / N, 1, T1n, 64);
        transpose_mat<<<grid1(64 * 512), 256, 0, stream>>>(rb[b].fW, 64, 512, Wt1);
        gemm(T1n, 64, Wt1, 64, nullptr, XWp, 512, N, 512, 64);
        xu_small<<<grid1((long)N * 8), 256, 0, stream>>>(T1n, 64, rb[b].fu, N, 64, Uf);
        fill_f32<<<grid1((long)N * 64), 256, 0, stream>>>(Sacc, (long)N * 64, 0.f);
        fill_f32<<<grid1(N), 256, 0, stream>>>(CNT, N, 0.f);
        feast_edge<<<grid1(E), 256, 0, stream>>>(Uf, rb[b].fc, XWp, src, tgt, E, Sacc, CNT);
        feast_edge<<<grid1(N), 256, 0, stream>>>(Uf, rb[b].fc, XWp, iota, iota, N, Sacc, CNT);
        feast_finish<<<grid1((long)N * 64), 256, 0, stream>>>(Sacc, CNT, rb[b].fb, N, T2);
        col_stats<<<64, 256, 0, stream>>>(T2, nullptr, N, 64, 64, 0, stats);
        bn_act<<<grid1((long)N * 64), 256, 0, stream>>>(
            T2, nullptr, stats, N, 64, 64, 0, 1.f / N, 1, T1n, 64);
        gemm(T1n, 64, rb[b].fc2W, 64, rb[b].fc2b, T3, 1024, N, 1024, 64);
        col_stats<<<1024, 256, 0, stream>>>(T3, HMAIN, N, 1024, 1024, 1024, stats);
        bn_act<<<grid1((long)N * 1024), 256, 0, stream>>>(
            T3, HMAIN, stats, N, 1024, 1024, 1024, 1.f / N, 1, HMAIN, 1024);
    }

    // ================= Head ================
    gemm(HMAIN, 1024, fc1W, 1024, fc1b, T1, 64, N, 64, 1024);
    col_stats<<<64, 256, 0, stream>>>(T1, nullptr, N, 64, 64, 0, stats);
    bn_act<<<grid1((long)N * 64), 256, 0, stream>>>(
        T1, nullptr, stats, N, 64, 64, 0, 1.f / N, 1, T1n, 64);
    gemm(T1n, 64, fc2W, 64, fc2b, (float*)d_out, 3, N, 3, 64);
}
